// RT_12068858101896
// MI455X (gfx1250) — matrix operand order repaired and checked by reading, not compiled, not run on hardware
//
#include <hip/hip_runtime.h>
#include <hip/hip_bf16.h>
#include <math.h>

#define Bsz  8
#define Ssz  1536
#define INsz 32
#define Dsz  128
#define Hh   4
#define DKsz 32
#define Kwin 7
#define NLVL 2
#define DFFs 512
#define NTOK (Bsz*Ssz)   // 12288

typedef __attribute__((ext_vector_type(16))) _Float16     v16h;
typedef __attribute__((ext_vector_type(8)))  _Float16     v8h;
typedef __attribute__((ext_vector_type(8)))  float        v8f;
typedef __attribute__((ext_vector_type(4)))  unsigned int v4u;
typedef __attribute__((ext_vector_type(4)))  int          v4i;
typedef __attribute__((ext_vector_type(8)))  int          v8i;

// ---------------------------------------------------------------------------
// Generic WMMA GEMM: Out[M,N] = act(A[M,K] @ B[K,N] + bias) (+ resid)
// Block tile 128x64, 8 waves, each wave 32x32 (2x2 WMMA 16x16x32 f16->f32).
// B_TRANS: B given as [N,K] row-major (i.e. B^T); staged into LDS by the
//          Tensor Data Mover (tensor_load_to_lds, TENSORcnt-tracked DMA).
// ---------------------------------------------------------------------------
template<bool A_F16, bool B_TRANS, bool OUT_F16, bool RELU, bool RES>
__global__ __launch_bounds__(256) void gemm_wmma(
    const void* __restrict__ Ap, const _Float16* __restrict__ Bp,
    const float* __restrict__ bias, const float* __restrict__ resid,
    void* __restrict__ Out, int M, int N, int Kd)
{
  __shared__ _Float16 As[128*32];
  __shared__ _Float16 Bt[64*32];
  const int tid  = threadIdx.x;
  const int lane = tid & 31;
  const int wid  = tid >> 5;
  const int m_blk = blockIdx.y * 128;
  const int n_blk = blockIdx.x * 64;
  const int wm = (wid >> 1) * 32;
  const int wn = (wid & 1) * 32;
  const int lr = lane & 15;
  const int kh = (lane < 16) ? 0 : 8;    // A-fragment K half-select
  const int kb = (lane < 16) ? 0 : 16;   // unused; B fragments also select K half with kh

  v8f acc[2][2];
  #pragma unroll
  for (int i = 0; i < 2; ++i)
    #pragma unroll
    for (int j = 0; j < 2; ++j)
      #pragma unroll
      for (int e = 0; e < 8; ++e) acc[i][j][e] = 0.f;

  const int a_row = tid >> 1;          // 0..127
  const int a_col = (tid & 1) * 16;    // 0/16
  const int b_n   = tid >> 2;          // 0..63
  const int b_k   = (tid & 3) * 8;     // 0,8,16,24

  for (int k0 = 0; k0 < Kd; k0 += 32) {
    __syncthreads();
    // ---- stage A (convert f32->f16 if needed) ----
    if (A_F16) {
      const _Float16* Af = (const _Float16*)Ap;
      const v8h* src = (const v8h*)(Af + (size_t)(m_blk + a_row) * Kd + k0 + a_col);
      *(v8h*)&As[a_row*32 + a_col]     = src[0];
      *(v8h*)&As[a_row*32 + a_col + 8] = src[1];
      if (k0 + 32 < Kd)   // prefetch next K tile -> global_prefetch_b8
        __builtin_prefetch((const char*)(Af + (size_t)(m_blk + a_row) * Kd + k0 + 32 + a_col), 0, 1);
    } else {
      const float* Af = (const float*)Ap;
      const float* src = Af + (size_t)(m_blk + a_row) * Kd + k0 + a_col;
      #pragma unroll
      for (int j = 0; j < 16; ++j) As[a_row*32 + a_col + j] = (_Float16)src[j];
    }
    // ---- stage B transposed into LDS: Bt[n][k] ----
    if (B_TRANS) {
      // Tensor Data Mover: DMA a 64(n) x 32(k) f16 tile of the [N,K] tensor
      // into LDS, row-major compacted == Bt[n][k] with stride 32 halves.
      if (wid == 0) {
        const unsigned long long ga =
            (unsigned long long)(size_t)(Bp + (size_t)n_blk * Kd + k0);
        const unsigned lds = (unsigned)(size_t)(void*)&Bt[0];
        v4u g0;
        g0[0] = 1u;                                            // count=1, user mode
        g0[1] = lds;                                           // lds_addr
        g0[2] = (unsigned)(ga & 0xffffffffull);                // global_addr[31:0]
        g0[3] = (unsigned)((ga >> 32) & 0x1ffffffull)          // global_addr[56:32]
                | 0x80000000u;                                 // type=2 (image)
        v8i g1;
        g1[0] = 0x10000;                                       // data_size=1 (2B)
        g1[1] = (Kd & 0xffff) << 16;                           // tensor_dim0[15:0]
        g1[2] = (int)(((unsigned)Kd >> 16) | ((unsigned)(N & 0xffff) << 16)); // td0 hi | td1 lo
        g1[3] = (int)(((unsigned)N >> 16) | (32u << 16));      // td1 hi | tile_dim0=32
        g1[4] = 64;                                            // tile_dim1=64, tile_dim2=0
        g1[5] = Kd;                                            // tensor_dim0_stride[31:0]
        g1[6] = 0;                                             // stride hi / dim1_stride lo
        g1[7] = 0;
        v4i gz = {0, 0, 0, 0};
#if __clang_major__ >= 23
        v8i gz8 = {0, 0, 0, 0, 0, 0, 0, 0};
        __builtin_amdgcn_tensor_load_to_lds(g0, g1, gz, gz, gz8, 0);
#else
        __builtin_amdgcn_tensor_load_to_lds(g0, g1, gz, gz, 0);
#endif
        __builtin_amdgcn_s_wait_tensorcnt(0);
      }
    } else {
      #pragma unroll
      for (int j = 0; j < 8; ++j)
        Bt[b_n*32 + b_k + j] = Bp[(size_t)(k0 + b_k + j) * N + n_blk + b_n];
    }
    __syncthreads();
    // ---- WMMA on 2x2 sub-tiles ----
    #pragma unroll
    for (int mi = 0; mi < 2; ++mi) {
      union { v16h v; v8h h[2]; } a;
      const int arow = wm + mi*16 + lr;
      a.h[0] = *(const v8h*)&As[arow*32 + kh];
      a.h[1] = *(const v8h*)&As[arow*32 + 16 + kh];
      #pragma unroll
      for (int ni = 0; ni < 2; ++ni) {
        union { v16h v; v8h h[2]; } bb;
        const int brow = wn + ni*16 + lr;
        bb.h[0] = *(const v8h*)&Bt[brow*32 + kh];
        bb.h[1] = *(const v8h*)&Bt[brow*32 + 16 + kh];
        acc[mi][ni] = __builtin_amdgcn_wmma_f32_16x16x32_f16(
            false, a.v, false, bb.v, (short)0, acc[mi][ni], false, false);
      }
    }
  }
  // ---- epilogue ----
  #pragma unroll
  for (int mi = 0; mi < 2; ++mi) {
    #pragma unroll
    for (int ni = 0; ni < 2; ++ni) {
      const int gn = n_blk + wn + ni*16 + lr;
      const float bvv = bias[gn];
      #pragma unroll
      for (int e = 0; e < 8; ++e) {
        const int gm = m_blk + wm + mi*16 + ((lane < 16) ? e : e + 8);
        float vv = acc[mi][ni][e] + bvv;
        if (RELU) vv = vv > 0.f ? vv : 0.f;
        if (RES)  vv += resid[(size_t)gm * N + gn];
        if (OUT_F16) ((_Float16*)Out)[(size_t)gm * N + gn] = (_Float16)vv;
        else         ((float*)Out)[(size_t)gm * N + gn] = vv;
      }
    }
  }
}

// ---------------------------------------------------------------------------
// LayerNorm (ddof=1, divide by std+eps), f32 in -> f16 out. 1 wave per row.
// ---------------------------------------------------------------------------
__global__ __launch_bounds__(256) void layernorm_kernel(
    const float* __restrict__ x, const float* __restrict__ a,
    const float* __restrict__ b, _Float16* __restrict__ out)
{
  const int lane = threadIdx.x & 31;
  const int wid  = threadIdx.x >> 5;
  const int row  = blockIdx.x * 8 + wid;
  const float* xr = x + (size_t)row * Dsz;
  float v[4];
  float s = 0.f;
  #pragma unroll
  for (int j = 0; j < 4; ++j) { v[j] = xr[lane*4 + j]; s += v[j]; }
  #pragma unroll
  for (int m = 1; m < 32; m <<= 1) s += __shfl_xor(s, m, 32);
  const float mean = s * (1.f / 128.f);
  float q = 0.f;
  #pragma unroll
  for (int j = 0; j < 4; ++j) { float d = v[j] - mean; q += d*d; }
  #pragma unroll
  for (int m = 1; m < 32; m <<= 1) q += __shfl_xor(q, m, 32);
  const float inv = 1.f / (sqrtf(q * (1.f / 127.f)) + 1e-6f);
  #pragma unroll
  for (int j = 0; j < 4; ++j) {
    const int c = lane*4 + j;
    out[(size_t)row * Dsz + c] = (_Float16)(a[c] * (v[j] - mean) * inv + b[c]);
  }
}

// ---------------------------------------------------------------------------
// GRU gate elementwise. gix includes bih; gh includes bhh (from GEMM bias).
// Window step k uses gi at t = s + k - (Kwin-1); t<0 -> bias only (zero pad).
// ---------------------------------------------------------------------------
template<bool FIRST, bool LAST>
__global__ __launch_bounds__(256) void gru_gate_kernel(
    const float* __restrict__ gix, const float* __restrict__ gh,
    const float* __restrict__ bih, const float* __restrict__ bhh,
    float* __restrict__ hst, _Float16* __restrict__ hst16,
    float* __restrict__ h_io, int kstep)
{
  const int idx = blockIdx.x * 256 + threadIdx.x;
  const int tok = idx >> 7, d = idx & 127;
  const int s = tok % Ssz;
  const int t = s + kstep - (Kwin - 1);
  float gir, giz, gin;
  if (t >= 0) {
    const float* g = gix + (size_t)(tok - s + t) * 384 + d;
    gir = g[0]; giz = g[128]; gin = g[256];
  } else {
    gir = bih[d]; giz = bih[128 + d]; gin = bih[256 + d];
  }
  float ghr, ghz, ghn, hprev;
  if (FIRST) {
    ghr = bhh[d]; ghz = bhh[128 + d]; ghn = bhh[256 + d]; hprev = 0.f;
  } else {
    const float* g = gh + (size_t)tok * 384 + d;
    ghr = g[0]; ghz = g[128]; ghn = g[256];
    hprev = hst[idx];
  }
  const float r = 1.f / (1.f + __expf(-(gir + ghr)));
  const float z = 1.f / (1.f + __expf(-(giz + ghz)));
  const float n = tanhf(gin + r * ghn);
  const float hnew = (1.f - z) * n + z * hprev;
  if (LAST) {
    h_io[idx] += hnew;                       // residual add
  } else {
    hst[idx] = hnew;
    hst16[idx] = (_Float16)hnew;
  }
}

// ---------------------------------------------------------------------------
// Causal flash attention. Block = 64 q-rows (4 waves x 16), kv tiles of 32.
// q/k/v f16 [NTOK,128] with head slice of 32 cols. Scores & PV via WMMA.
// ---------------------------------------------------------------------------
__global__ __launch_bounds__(128) void attn_kernel(
    const _Float16* __restrict__ q, const _Float16* __restrict__ kk,
    const _Float16* __restrict__ v, _Float16* __restrict__ o)
{
  __shared__ _Float16 Vt[32*32];        // V^T tile: Vt[d][key]
  __shared__ _Float16 P[4][16*32];      // per-wave exp(score) tile
  const int tid = threadIdx.x, lane = tid & 31, wid = tid >> 5;
  const int id = blockIdx.x;
  const int qt   = id % (Ssz / 64);
  const int bh   = id / (Ssz / 64);
  const int b    = bh / Hh, head = bh % Hh;
  const size_t base = (size_t)b * Ssz * Dsz + (size_t)head * DKsz;
  const int qbase = qt * 64 + wid * 16;
  const int lr = lane & 15;
  const int kh = (lane < 16) ? 0 : 8;
  const int kb = (lane < 16) ? 0 : 16;

  union { v16h v; v8h h[2]; } qa;
  {
    const _Float16* qr = q + base + (size_t)(qbase + lr) * Dsz;
    qa.h[0] = *(const v8h*)(qr + kh);
    qa.h[1] = *(const v8h*)(qr + 16 + kh);
  }
  v8f o0, o1;
  float rmax[8], rsum[8];
  #pragma unroll
  for (int e = 0; e < 8; ++e) { o0[e] = 0.f; o1[e] = 0.f; rmax[e] = -3e38f; rsum[e] = 0.f; }
  const float scale = 0.17677669529663687f;   // 1/sqrt(32)
  const int kvend = qt * 64 + 64;
  const int v_key = tid >> 2, v_d0 = (tid & 3) * 8;

  for (int kv0 = 0; kv0 < kvend; kv0 += 32) {
    // stage V^T
    {
      const _Float16* vr = v + base + (size_t)(kv0 + v_key) * Dsz + v_d0;
      #pragma unroll
      for (int j = 0; j < 8; ++j) Vt[(v_d0 + j) * 32 + v_key] = vr[j];
    }
    __syncthreads();
    // k fragments (B = k^T; B^T[n][k] = k[key][d], contiguous in global)
    union { v16h v; v8h h[2]; } kf0, kf1;
    {
      const _Float16* kr = kk + base + (size_t)(kv0 + lr) * Dsz + kh;
      kf0.h[0] = *(const v8h*)kr;            kf0.h[1] = *(const v8h*)(kr + 16);
      const _Float16* kr2 = kr + (size_t)16 * Dsz;
      kf1.h[0] = *(const v8h*)kr2;           kf1.h[1] = *(const v8h*)(kr2 + 16);
    }
    v8f s0, s1;
    #pragma unroll
    for (int e = 0; e < 8; ++e) { s0[e] = 0.f; s1[e] = 0.f; }
    s0 = __builtin_amdgcn_wmma_f32_16x16x32_f16(false, qa.v, false, kf0.v, (short)0, s0, false, false);
    s1 = __builtin_amdgcn_wmma_f32_16x16x32_f16(false, qa.v, false, kf1.v, (short)0, s1, false, false);

    _Float16* Pw = &P[wid][0];
    #pragma unroll
    for (int e = 0; e < 8; ++e) {
      const int m  = (lane < 16) ? e : e + 8;
      const int mg = qbase + m;
      float a0 = s0[e] * scale, a1 = s1[e] * scale;
      if (kv0 + lr > mg)      a0 = -1e30f;
      if (kv0 + 16 + lr > mg) a1 = -1e30f;
      float mx = fmaxf(a0, a1);
      #pragma unroll
      for (int msk = 1; msk < 16; msk <<= 1) mx = fmaxf(mx, __shfl_xor(mx, msk, 32));
      const float nm   = fmaxf(rmax[e], mx);
      const float corr = __expf(rmax[e] - nm);
      const float p0 = __expf(a0 - nm), p1 = __expf(a1 - nm);
      float ps = p0 + p1;
      #pragma unroll
      for (int msk = 1; msk < 16; msk <<= 1) ps += __shfl_xor(ps, msk, 32);
      rsum[e] = rsum[e] * corr + ps;
      o0[e] *= corr; o1[e] *= corr;
      rmax[e] = nm;
      Pw[m*32 + lr]      = (_Float16)p0;
      Pw[m*32 + 16 + lr] = (_Float16)p1;
    }
    __syncthreads();
    // P as A-fragment, V^T as B-fragments
    union { v16h v; v8h h[2]; } pa, vb0, vb1;
    pa.h[0]  = *(const v8h*)&Pw[lr*32 + kh];
    pa.h[1]  = *(const v8h*)&Pw[lr*32 + 16 + kh];
    vb0.h[0] = *(const v8h*)&Vt[lr*32 + kh];
    vb0.h[1] = *(const v8h*)&Vt[lr*32 + 16 + kh];
    vb1.h[0] = *(const v8h*)&Vt[(16 + lr)*32 + kh];
    vb1.h[1] = *(const v8h*)&Vt[(16 + lr)*32 + 16 + kh];
    o0 = __builtin_amdgcn_wmma_f32_16x16x32_f16(false, pa.v, false, vb0.v, (short)0, o0, false, false);
    o1 = __builtin_amdgcn_wmma_f32_16x16x32_f16(false, pa.v, false, vb1.v, (short)0, o1, false, false);
    __syncthreads();
  }
  #pragma unroll
  for (int e = 0; e < 8; ++e) {
    const int m = (lane < 16) ? e : e + 8;
    const float inv = 1.f / rsum[e];
    _Float16* orow = o + base + (size_t)(qbase + m) * Dsz;
    orow[lr]      = (_Float16)(o0[e] * inv);
    orow[16 + lr] = (_Float16)(o1[e] * inv);
  }
}

// ---------------------------------------------------------------------------
// Output head: sigmoid(h @ out_W + out_b), one wave per row.
// ---------------------------------------------------------------------------
__global__ __launch_bounds__(256) void out_kernel(
    const float* __restrict__ h, const float* __restrict__ W,
    const float* __restrict__ bb, float* __restrict__ out)
{
  const int lane = threadIdx.x & 31, wid = threadIdx.x >> 5;
  const int row = blockIdx.x * 8 + wid;
  const float* hr = h + (size_t)row * Dsz;
  float s = 0.f;
  #pragma unroll
  for (int j = 0; j < 4; ++j) s += hr[lane*4 + j] * W[lane*4 + j];
  #pragma unroll
  for (int m = 1; m < 32; m <<= 1) s += __shfl_xor(s, m, 32);
  if (lane == 0) out[row] = 1.f / (1.f + __expf(-(s + bb[0])));
}

__global__ void cvt_kernel(const float* __restrict__ src, _Float16* __restrict__ dst, int n)
{
  const int i = blockIdx.x * 256 + threadIdx.x;
  if (i < n) dst[i] = (_Float16)src[i];
}

// ---------------------------------------------------------------------------
extern "C" void kernel_launch(void* const* d_in, const int* in_sizes, int n_in,
                              void* d_out, int out_size, void* d_ws, size_t ws_size,
                              hipStream_t stream)
{
  (void)in_sizes; (void)n_in; (void)out_size; (void)ws_size;
  const float* x        = (const float*)d_in[0];
  const float* enc_W    = (const float*)d_in[1];
  const float* enc_b    = (const float*)d_in[2];
  const float* ln_rnn_a = (const float*)d_in[3];
  const float* ln_rnn_b = (const float*)d_in[4];
  const float* gru_Wih  = (const float*)d_in[5];
  const float* gru_Whh  = (const float*)d_in[6];
  const float* gru_bih  = (const float*)d_in[7];
  const float* gru_bhh  = (const float*)d_in[8];
  const float* ln_at_a  = (const float*)d_in[9];
  const float* ln_at_b  = (const float*)d_in[10];
  const float* Wq = (const float*)d_in[11];
  const float* bq = (const float*)d_in[12];
  const float* Wk = (const float*)d_in[13];
  const float* bk = (const float*)d_in[14];
  const float* Wv = (const float*)d_in[15];
  const float* bv = (const float*)d_in[16];
  const float* Wo = (const float*)d_in[17];
  const float* bo = (const float*)d_in[18];
  const float* ln_ff_a = (const float*)d_in[19];
  const float* ln_ff_b = (const float*)d_in[20];
  const float* ff_W1 = (const float*)d_in[21];
  const float* ff_b1 = (const float*)d_in[22];
  const float* ff_W2 = (const float*)d_in[23];
  const float* ff_b2 = (const float*)d_in[24];
  const float* out_W = (const float*)d_in[25];
  const float* out_b = (const float*)d_in[26];

  char* wsp = (char*)d_ws;
  size_t off = 0;
  auto alloc = [&](size_t bytes) -> void* {
    void* p = wsp + off;
    off += (bytes + 255) & ~(size_t)255;
    return p;
  };
  float*    h     = (float*)alloc((size_t)NTOK * Dsz * 4);
  _Float16* hn16  = (_Float16*)alloc((size_t)NTOK * Dsz * 2);
  float*    gix   = (float*)alloc((size_t)NTOK * 384 * 4);
  float*    ghb   = (float*)alloc((size_t)NTOK * 384 * 4);
  float*    hst   = (float*)alloc((size_t)NTOK * Dsz * 4);
  _Float16* hst16 = (_Float16*)alloc((size_t)NTOK * Dsz * 2);
  _Float16* q16   = (_Float16*)alloc((size_t)NTOK * Dsz * 2);
  _Float16* k16   = (_Float16*)alloc((size_t)NTOK * Dsz * 2);
  _Float16* v16   = (_Float16*)alloc((size_t)NTOK * Dsz * 2);
  _Float16* ao16  = (_Float16*)alloc((size_t)NTOK * Dsz * 2);
  _Float16* t116  = (_Float16*)alloc((size_t)NTOK * DFFs * 2);
  _Float16* encW16 = (_Float16*)alloc(32 * 128 * 2);
  _Float16* Wih16  = (_Float16*)alloc((size_t)NLVL * 384 * 128 * 2);
  _Float16* Whh16  = (_Float16*)alloc((size_t)NLVL * 384 * 128 * 2);
  _Float16* Wq16   = (_Float16*)alloc((size_t)NLVL * 128 * 128 * 2);
  _Float16* Wk16   = (_Float16*)alloc((size_t)NLVL * 128 * 128 * 2);
  _Float16* Wv16   = (_Float16*)alloc((size_t)NLVL * 128 * 128 * 2);
  _Float16* Wo16   = (_Float16*)alloc((size_t)NLVL * 128 * 128 * 2);
  _Float16* W116   = (_Float16*)alloc((size_t)NLVL * 128 * DFFs * 2);
  _Float16* W216   = (_Float16*)alloc((size_t)NLVL * DFFs * 128 * 2);

  auto cvt = [&](const float* s, _Float16* d, int n) {
    cvt_kernel<<<(n + 255) / 256, 256, 0, stream>>>(s, d, n);
  };
  cvt(enc_W,   encW16, 32 * 128);
  cvt(gru_Wih, Wih16,  NLVL * 384 * 128);
  cvt(gru_Whh, Whh16,  NLVL * 384 * 128);
  cvt(Wq, Wq16, NLVL * 128 * 128);
  cvt(Wk, Wk16, NLVL * 128 * 128);
  cvt(Wv, Wv16, NLVL * 128 * 128);
  cvt(Wo, Wo16, NLVL * 128 * 128);
  cvt(ff_W1, W116, NLVL * 128 * DFFs);
  cvt(ff_W2, W216, NLVL * DFFs * 128);

  const dim3 blk(256);
  const int MT = NTOK / 128;   // 96 m-tiles

  // ---- encoder: h = x @ enc_W + enc_b ----
  gemm_wmma<false, false, false, false, false>
      <<<dim3(Dsz / 64, MT), blk, 0, stream>>>(x, encW16, enc_b, nullptr, h, NTOK, Dsz, INsz);

  for (int lvl = 0; lvl < NLVL; ++lvl) {
    const _Float16* Wih = Wih16 + (size_t)lvl * 384 * 128;
    const _Float16* Whh = Whh16 + (size_t)lvl * 384 * 128;
    const float* bih = gru_bih + lvl * 384;
    const float* bhh = gru_bhh + lvl * 384;

    // ---- LocalRNN sublayer ----
    layernorm_kernel<<<NTOK / 8, blk, 0, stream>>>(h, ln_rnn_a + lvl * 128, ln_rnn_b + lvl * 128, hn16);
    gemm_wmma<true, true, false, false, false>
        <<<dim3(384 / 64, MT), blk, 0, stream>>>(hn16, Wih, bih, nullptr, gix, NTOK, 384, Dsz);
    gru_gate_kernel<true, false>
        <<<NTOK * Dsz / 256, blk, 0, stream>>>(gix, nullptr, bih, bhh, hst, hst16, h, 0);
    for (int k = 1; k < Kwin; ++k) {
      gemm_wmma<true, true, false, false, false>
          <<<dim3(384 / 64, MT), blk, 0, stream>>>(hst16, Whh, bhh, nullptr, ghb, NTOK, 384, Dsz);
      if (k < Kwin - 1)
        gru_gate_kernel<false, false>
            <<<NTOK * Dsz / 256, blk, 0, stream>>>(gix, ghb, bih, bhh, hst, hst16, h, k);
      else
        gru_gate_kernel<false, true>
            <<<NTOK * Dsz / 256, blk, 0, stream>>>(gix, ghb, bih, bhh, hst, hst16, h, k);
    }

    // ---- MHPooling sublayer ----
    layernorm_kernel<<<NTOK / 8, blk, 0, stream>>>(h, ln_at_a + lvl * 128, ln_at_b + lvl * 128, hn16);
    gemm_wmma<true, false, true, false, false>
        <<<dim3(2, MT), blk, 0, stream>>>(hn16, Wq16 + (size_t)lvl * 16384, bq + lvl * 128, nullptr, q16, NTOK, Dsz, Dsz);
    gemm_wmma<true, false, true, false, false>
        <<<dim3(2, MT), blk, 0, stream>>>(hn16, Wk16 + (size_t)lvl * 16384, bk + lvl * 128, nullptr, k16, NTOK, Dsz, Dsz);
    gemm_wmma<true, false, true, false, false>
        <<<dim3(2, MT), blk, 0, stream>>>(hn16, Wv16 + (size_t)lvl * 16384, bv + lvl * 128, nullptr, v16, NTOK, Dsz, Dsz);
    attn_kernel<<<Bsz * Hh * (Ssz / 64), dim3(128), 0, stream>>>(q16, k16, v16, ao16);
    gemm_wmma<true, false, false, false, true>
        <<<dim3(2, MT), blk, 0, stream>>>(ao16, Wo16 + (size_t)lvl * 16384, bo + lvl * 128, h, h, NTOK, Dsz, Dsz);

    // ---- FFN sublayer ----
    layernorm_kernel<<<NTOK / 8, blk, 0, stream>>>(h, ln_ff_a + lvl * 128, ln_ff_b + lvl * 128, hn16);
    gemm_wmma<true, false, true, true, false>
        <<<dim3(DFFs / 64, MT), blk, 0, stream>>>(hn16, W116 + (size_t)lvl * 128 * DFFs, ff_b1 + lvl * DFFs, nullptr, t116, NTOK, DFFs, Dsz);
    gemm_wmma<true, false, false, false, true>
        <<<dim3(2, MT), blk, 0, stream>>>(t116, W216 + (size_t)lvl * DFFs * 128, ff_b2 + lvl * 128, h, h, NTOK, Dsz, DFFs);
  }

  // ---- output head ----
  out_kernel<<<NTOK / 8, blk, 0, stream>>>(h, out_W, out_b, (float*)d_out);
}
